// DecoderNetwork_27049704030336
// MI455X (gfx1250) — matrix operand order repaired and checked by reading, not compiled, not run on hardware
//
#include <hip/hip_runtime.h>
#include <hip/hip_bf16.h>
#include <math.h>

// ---------------------------------------------------------------------------
// Problem constants (B, T, D, H fixed by the reference)
// ---------------------------------------------------------------------------
#define BB   2
#define TT   1024
#define DD   1024
#define HH   16
#define HS   64
#define MTOT (BB * TT)          // 2048 rows of activations
#define SCALE_Q 8.0f            // sqrt(HS); reference multiplies scores by it

typedef __attribute__((ext_vector_type(16))) __bf16 v16bf;
typedef __attribute__((ext_vector_type(8)))  float  f32x8;
typedef unsigned short u16;

union FragAB {
    v16bf v;
    uint4 u[2];
};

__device__ __forceinline__ u16 f2bf(float f) {
    unsigned int u = __float_as_uint(f);
    u += 0x7FFFu + ((u >> 16) & 1u);     // round-to-nearest-even
    return (u16)(u >> 16);
}

__device__ __forceinline__ f32x8 wmma_bf16(const FragAB& a, const FragAB& b, f32x8 c) {
    // v_wmma_f32_16x16x32_bf16  (D = A x B + C, f32 accumulate)
    return __builtin_amdgcn_wmma_f32_16x16x32_bf16(
        /*neg_a=*/false, a.v, /*neg_b=*/false, b.v,
        /*c_mod=*/(short)0, c, /*reuse_a=*/false, /*reuse_b=*/false);
}

// ---------------------------------------------------------------------------
// RMSNorm: per row of D=1024, fp32 in -> bf16 out.  block = 256 threads
// ---------------------------------------------------------------------------
__global__ void rmsnorm_kernel(const float* __restrict__ x,
                               const float* __restrict__ w,
                               u16* __restrict__ y) {
    int row = blockIdx.x;
    const float* xr = x + (size_t)row * DD;
    float v[4];
    float s = 0.f;
#pragma unroll
    for (int i = 0; i < 4; ++i) {
        v[i] = xr[threadIdx.x + i * 256];
        s += v[i] * v[i];
    }
#pragma unroll
    for (int m = 16; m >= 1; m >>= 1) s += __shfl_xor(s, m, 32);
    __shared__ float red[8];
    if ((threadIdx.x & 31) == 0) red[threadIdx.x >> 5] = s;
    __syncthreads();
    float tot = red[0];
#pragma unroll
    for (int i = 1; i < 8; ++i) tot += red[i];
    float r = rsqrtf(tot * (1.0f / DD) + 1e-5f);
#pragma unroll
    for (int i = 0; i < 4; ++i) {
        int c = threadIdx.x + i * 256;
        y[(size_t)row * DD + c] = f2bf(v[i] * r * w[c]);
    }
}

// ---------------------------------------------------------------------------
// Weight converters -> bf16, laid out [N][K] so WMMA B-fragments are
// contiguous b128 loads.
// ---------------------------------------------------------------------------
// head weight (H, D, HS) -> wt[n*D + k], n = h*HS + e
__global__ void conv_head_w(const float* __restrict__ w, u16* __restrict__ wt) {
    __shared__ float tile[32][65];
    int h  = blockIdx.y;
    int k0 = blockIdx.x * 32;
    int tx = threadIdx.x & 63, ty = threadIdx.x >> 6;     // 256 thr: 4 rows/iter
    for (int i = ty; i < 32; i += 4)
        tile[i][tx] = w[((size_t)h * DD + (k0 + i)) * HS + tx];
    __syncthreads();
    int kk = threadIdx.x & 31, ee = threadIdx.x >> 5;     // 8 e-rows/iter
    for (int e = ee; e < HS; e += 8)
        wt[((size_t)h * HS + e) * DD + k0 + kk] = f2bf(tile[kk][e]);
}

// plain (K, N) -> wt[n*K + k]   (tiled transpose)
__global__ void conv_plain_w(const float* __restrict__ w, u16* __restrict__ wt,
                             int K, int N) {
    __shared__ float tile[32][33];
    int k0 = blockIdx.x * 32, n0 = blockIdx.y * 32;
    int tx = threadIdx.x & 31, ty = threadIdx.x >> 5;     // 256 thr: 8 rows/iter
    for (int i = ty; i < 32; i += 8)
        tile[i][tx] = w[(size_t)(k0 + i) * N + (n0 + tx)];
    __syncthreads();
    for (int i = ty; i < 32; i += 8)
        wt[(size_t)(n0 + i) * K + (k0 + tx)] = f2bf(tile[tx][i]);
}

// V [B*T][D] (col = h*HS+f) -> Vt [(b*H+h)*HS + f][T]  so the PV B-fragment
// (fixed feature, contiguous keys) is a pair of b128 loads.
__global__ void transpose_v(const u16* __restrict__ v, u16* __restrict__ vt) {
    __shared__ u16 tile[32][66];
    int bh = blockIdx.y;
    int b = bh >> 4, h = bh & 15;
    int t0 = blockIdx.x * 32;
    int tx = threadIdx.x & 63, ty = threadIdx.x >> 6;
    for (int i = ty; i < 32; i += 4)
        tile[i][tx] = v[(size_t)(b * TT + t0 + i) * DD + h * HS + tx];
    __syncthreads();
    int kk = threadIdx.x & 31, ee = threadIdx.x >> 5;
    for (int f = ee; f < HS; f += 8)
        vt[((size_t)bh * HS + f) * TT + t0 + kk] = tile[kk][f];
}

// ---------------------------------------------------------------------------
// WMMA GEMM: C[M][N] = epilogue( A[M][K](bf16) * Wt[N][K](bf16) )
// One wave per 32x64 tile; 2x4 fragment grid; 8 v_wmma per K=32 step from
// 6 fragment loads (24 B of L2 traffic per WMMA).  Sized to stay well under
// the 256 directly-addressable VGPRs (64 acc + ~96 pipelined fragments).
// Fragment layouts per CDNA5 ISA:
//   A lane(l,hi): row M=l; u[0]=K k0+hi*8..+7, u[1]=K k0+16+hi*8..+7
//   B lane(l,hi): col N=l; u[0]=K k0+hi*8..+7, u[1]=K k0+16+hi*8..+7
//   D elem r:     row M=hi*8+r, col N=l
// ---------------------------------------------------------------------------
template <bool BIAS, bool GELU_, bool RES, bool OUTF, bool OUTB>
__global__ void gemm_wmma(const u16* __restrict__ A, const u16* __restrict__ Wt,
                          const float* __restrict__ bias,
                          const float* __restrict__ res,
                          float* __restrict__ outf, u16* __restrict__ outb,
                          int N, int K, float alpha) {
    int m0 = blockIdx.x * 32;
    int n0 = blockIdx.y * 64;
    int lane = threadIdx.x;
    int l = lane & 15, hi = lane >> 4;

    const f32x8 z8 = {0.f, 0.f, 0.f, 0.f, 0.f, 0.f, 0.f, 0.f};
    f32x8 acc[2][4];
#pragma unroll
    for (int mi = 0; mi < 2; ++mi)
#pragma unroll
        for (int ni = 0; ni < 4; ++ni) acc[mi][ni] = z8;

    const u16* a0 = A + (size_t)(m0 + l)      * K + hi * 8;
    const u16* a1 = A + (size_t)(m0 + 16 + l) * K + hi * 8;
    const u16* bp[4];
#pragma unroll
    for (int ni = 0; ni < 4; ++ni)
        bp[ni] = Wt + (size_t)(n0 + ni * 16 + l) * K + hi * 8;

#pragma unroll 2
    for (int k0 = 0; k0 < K; k0 += 32) {
        if (k0 + 64 < K) {
            __builtin_prefetch(a0 + k0 + 64, 0, 1);      // global_prefetch_b8
            __builtin_prefetch(bp[0] + k0 + 64, 0, 1);
            __builtin_prefetch(bp[2] + k0 + 64, 0, 1);
        }
        FragAB fa0, fa1, fb[4];
        fa0.u[0] = *(const uint4*)(a0 + k0);
        fa0.u[1] = *(const uint4*)(a0 + k0 + 16);
        fa1.u[0] = *(const uint4*)(a1 + k0);
        fa1.u[1] = *(const uint4*)(a1 + k0 + 16);
#pragma unroll
        for (int ni = 0; ni < 4; ++ni) {
            fb[ni].u[0] = *(const uint4*)(bp[ni] + k0);
            fb[ni].u[1] = *(const uint4*)(bp[ni] + k0 + 16);
        }
#pragma unroll
        for (int ni = 0; ni < 4; ++ni) {
            acc[0][ni] = wmma_bf16(fa0, fb[ni], acc[0][ni]);
            acc[1][ni] = wmma_bf16(fa1, fb[ni], acc[1][ni]);
        }
    }

#pragma unroll
    for (int ni = 0; ni < 4; ++ni) {
        int col = n0 + ni * 16 + l;
        float bv = BIAS ? bias[col] : 0.0f;
#pragma unroll
        for (int mi = 0; mi < 2; ++mi) {
#pragma unroll
            for (int r = 0; r < 8; ++r) {
                int row = m0 + mi * 16 + hi * 8 + r;
                float t = (acc[mi][ni][r] + bv) * alpha;
                if (GELU_) t = 0.5f * t * (1.0f + erff(t * 0.7071067811865476f));
                if (RES)  t += res[(size_t)row * N + col];
                if (OUTF) outf[(size_t)row * N + col] = t;
                if (OUTB) outb[(size_t)row * N + col] = f2bf(t);
            }
        }
    }
}

// ---------------------------------------------------------------------------
// Flash attention, one wave per (b, h, 32-query tile).
// Q already holds scale (and q-bias for the final attn).  Scores f32,
// online softmax with 16-lane shfl_xor row reductions, P re-shaped
// C/D->A layout through a 2KB LDS tile, PV via WMMA from pre-transposed V.
// ---------------------------------------------------------------------------
template <bool CAUSAL>
__global__ void attn_wmma(const u16* __restrict__ Qb, const u16* __restrict__ Kb,
                          const u16* __restrict__ Vt, u16* __restrict__ Ob) {
    __shared__ u16 pls[32 * 32];
    int bh = blockIdx.y;
    int b = bh >> 4, h = bh & 15;
    int qt0 = blockIdx.x * 32;
    int lane = threadIdx.x, l = lane & 15, hi = lane >> 4;

    // Q A-fragments, kept in registers for the whole key loop
    FragAB qa[2][2];
#pragma unroll
    for (int mi = 0; mi < 2; ++mi) {
        const u16* qp = Qb + (size_t)(b * TT + qt0 + mi * 16 + l) * DD + h * HS;
#pragma unroll
        for (int ks = 0; ks < 2; ++ks) {
            qa[mi][ks].u[0] = *(const uint4*)(qp + ks * 32 + hi * 8);
            qa[mi][ks].u[1] = *(const uint4*)(qp + ks * 32 + 16 + hi * 8);
        }
    }

    const f32x8 z8 = {0.f, 0.f, 0.f, 0.f, 0.f, 0.f, 0.f, 0.f};
    f32x8 oacc[2][4] = {{z8, z8, z8, z8}, {z8, z8, z8, z8}};
    float m_run[2][8], l_run[2][8];
#pragma unroll
    for (int mi = 0; mi < 2; ++mi)
#pragma unroll
        for (int r = 0; r < 8; ++r) { m_run[mi][r] = -1e30f; l_run[mi][r] = 0.f; }

    int nst = CAUSAL ? ((int)blockIdx.x + 1) : (TT / 32);
    for (int st = 0; st < nst; ++st) {
        int s0 = st * 32;

        // ---- S = Q K^T  (2x2 tiles, K-dim = HS = 2 steps of 32) ----
        f32x8 sacc[2][2] = {{z8, z8}, {z8, z8}};
#pragma unroll
        for (int ks = 0; ks < 2; ++ks) {
            FragAB kb[2];
#pragma unroll
            for (int ni = 0; ni < 2; ++ni) {
                const u16* kp = Kb + (size_t)(b * TT + s0 + ni * 16 + l) * DD
                                   + h * HS + ks * 32 + hi * 8;
                kb[ni].u[0] = *(const uint4*)(kp);
                kb[ni].u[1] = *(const uint4*)(kp + 16);
            }
#pragma unroll
            for (int mi = 0; mi < 2; ++mi)
#pragma unroll
                for (int ni = 0; ni < 2; ++ni)
                    sacc[mi][ni] = wmma_bf16(qa[mi][ks], kb[ni], sacc[mi][ni]);
        }

        if (CAUSAL && st == (int)blockIdx.x) {
#pragma unroll
            for (int mi = 0; mi < 2; ++mi)
#pragma unroll
                for (int ni = 0; ni < 2; ++ni)
#pragma unroll
                    for (int r = 0; r < 8; ++r) {
                        int q = qt0 + mi * 16 + hi * 8 + r;
                        int kk = s0 + ni * 16 + l;
                        if (kk > q) sacc[mi][ni][r] = -1e30f;
                    }
        }

        // ---- online softmax update ----
#pragma unroll
        for (int mi = 0; mi < 2; ++mi)
#pragma unroll
            for (int r = 0; r < 8; ++r) {
                float mx = fmaxf(sacc[mi][0][r], sacc[mi][1][r]);
#pragma unroll
                for (int d = 1; d < 16; d <<= 1) mx = fmaxf(mx, __shfl_xor(mx, d, 32));
                float mnew = fmaxf(m_run[mi][r], mx);
                float f = __expf(m_run[mi][r] - mnew);
                float p0 = __expf(sacc[mi][0][r] - mnew);
                float p1 = __expf(sacc[mi][1][r] - mnew);
                sacc[mi][0][r] = p0;
                sacc[mi][1][r] = p1;
                float rs = p0 + p1;
#pragma unroll
                for (int d = 1; d < 16; d <<= 1) rs += __shfl_xor(rs, d, 32);
                l_run[mi][r] = l_run[mi][r] * f + rs;
                m_run[mi][r] = mnew;
#pragma unroll
                for (int nf = 0; nf < 4; ++nf) oacc[mi][nf][r] *= f;
            }

        // ---- P (C/D layout) -> LDS -> A-layout fragments ----
#pragma unroll
        for (int mi = 0; mi < 2; ++mi)
#pragma unroll
            for (int ni = 0; ni < 2; ++ni)
#pragma unroll
                for (int r = 0; r < 8; ++r)
                    pls[(mi * 16 + hi * 8 + r) * 32 + ni * 16 + l] =
                        f2bf(sacc[mi][ni][r]);
        __syncthreads();

        FragAB pa[2];
#pragma unroll
        for (int mi = 0; mi < 2; ++mi) {
            const u16* pp = pls + (size_t)(mi * 16 + l) * 32;
            pa[mi].u[0] = *(const uint4*)(pp + hi * 8);
            pa[mi].u[1] = *(const uint4*)(pp + 16 + hi * 8);
        }

        // ---- O += P * V ----
#pragma unroll
        for (int nf = 0; nf < 4; ++nf) {
            const u16* vp = Vt + (size_t)(bh * HS + nf * 16 + l) * TT + s0 + hi * 8;
            FragAB vb;
            vb.u[0] = *(const uint4*)(vp);
            vb.u[1] = *(const uint4*)(vp + 16);
#pragma unroll
            for (int mi = 0; mi < 2; ++mi)
                oacc[mi][nf] = wmma_bf16(pa[mi], vb, oacc[mi][nf]);
        }
        __syncthreads();
    }

    // ---- epilogue: normalize, store bf16, heads concatenated ----
#pragma unroll
    for (int mi = 0; mi < 2; ++mi)
#pragma unroll
        for (int r = 0; r < 8; ++r) {
            float inv = 1.0f / l_run[mi][r];
            size_t row = (size_t)(b * TT + qt0 + mi * 16 + hi * 8 + r);
#pragma unroll
            for (int nf = 0; nf < 4; ++nf)
                Ob[row * DD + h * HS + nf * 16 + l] = f2bf(oacc[mi][nf][r] * inv);
        }
}

// ---------------------------------------------------------------------------
// Host-side orchestration
// ---------------------------------------------------------------------------
extern "C" void kernel_launch(void* const* d_in, const int* in_sizes, int n_in,
                              void* d_out, int out_size, void* d_ws, size_t ws_size,
                              hipStream_t stream) {
    const float* src  = (const float*)d_in[0];
    const float* att  = (const float*)d_in[1];
    const float* wq_m = (const float*)d_in[2];
    const float* wk_m = (const float*)d_in[3];
    const float* wv_m = (const float*)d_in[4];
    const float* pm_w = (const float*)d_in[5];
    const float* pm_b = (const float*)d_in[6];
    const float* wq_f = (const float*)d_in[7];
    const float* wk_f = (const float*)d_in[8];
    const float* wv_f = (const float*)d_in[9];
    const float* bq_f = (const float*)d_in[10];
    const float* bk_f = (const float*)d_in[11];
    const float* bv_f = (const float*)d_in[12];
    const float* pf_w = (const float*)d_in[13];
    const float* pf_b = (const float*)d_in[14];
    const float* w1   = (const float*)d_in[15];
    const float* b1   = (const float*)d_in[16];
    const float* w2   = (const float*)d_in[17];
    const float* b2   = (const float*)d_in[18];
    const float* nw   = (const float*)d_in[19];
    float* out = (float*)d_out;

    char* p = (char*)d_ws;
    auto take = [&](size_t bytes) -> char* {
        char* r = p;
        p += (bytes + 255) & ~(size_t)255;
        return r;
    };
    const size_t DD2 = (size_t)DD * DD;           // 1M elements
    const size_t MD  = (size_t)MTOT * DD;         // 2M elements

    // bf16 weights, [N][K]
    u16* Wqm = (u16*)take(DD2 * 2);
    u16* Wkm = (u16*)take(DD2 * 2);
    u16* Wvm = (u16*)take(DD2 * 2);
    u16* Wqf = (u16*)take(DD2 * 2);
    u16* Wkf = (u16*)take(DD2 * 2);
    u16* Wvf = (u16*)take(DD2 * 2);
    u16* Pm  = (u16*)take(DD2 * 2);
    u16* Pf  = (u16*)take(DD2 * 2);
    u16* W1t = (u16*)take(DD2 * 4 * 2);           // [4096][1024]
    u16* W2t = (u16*)take(DD2 * 4 * 2);           // [1024][4096]
    // activations (reused across phases)
    u16*   Xn    = (u16*)take(MD * 2);            // norm(src) / norm(att) / norm(f_out)
    u16*   Qb    = (u16*)take(MD * 2);
    u16*   Kbf   = (u16*)take(MD * 2);
    u16*   Vb    = (u16*)take(MD * 2);
    u16*   Vtb   = (u16*)take(MD * 2);
    u16*   Ob    = (u16*)take(MD * 2);
    float* moutF = (float*)take(MD * 4);
    u16*   moutB = (u16*)take(MD * 2);
    float* foutF = (float*)take(MD * 4);
    u16*   H1    = (u16*)take(MD * 4 * 2);        // [2048][4096]

    dim3 bConv(256);
    dim3 gHead(DD / 32, HH);
    // weight conversion (cheap: ~67MB read once, L2-resident afterwards)
    conv_head_w<<<gHead, bConv, 0, stream>>>(wq_m, Wqm);
    conv_head_w<<<gHead, bConv, 0, stream>>>(wk_m, Wkm);
    conv_head_w<<<gHead, bConv, 0, stream>>>(wv_m, Wvm);
    conv_head_w<<<gHead, bConv, 0, stream>>>(wq_f, Wqf);
    conv_head_w<<<gHead, bConv, 0, stream>>>(wk_f, Wkf);
    conv_head_w<<<gHead, bConv, 0, stream>>>(wv_f, Wvf);
    conv_plain_w<<<dim3(32, 32), bConv, 0, stream>>>(pm_w, Pm, DD, DD);
    conv_plain_w<<<dim3(32, 32), bConv, 0, stream>>>(pf_w, Pf, DD, DD);
    conv_plain_w<<<dim3(32, 128), bConv, 0, stream>>>(w1, W1t, DD, 4 * DD);
    conv_plain_w<<<dim3(128, 32), bConv, 0, stream>>>(w2, W2t, 4 * DD, DD);

    dim3 gN1(MTOT / 32, DD / 64);       // GEMM grids (32x64 tiles), N = 1024
    dim3 gN4(MTOT / 32, 4 * DD / 64);   // N = 4096
    dim3 bW(32);                        // one wave per tile
    dim3 gAtt(TT / 32, BB * HH);

    // ---- masked self-attention block ----
    rmsnorm_kernel<<<MTOT, 256, 0, stream>>>(src, nw, Xn);
    gemm_wmma<false, false, false, false, true><<<gN1, bW, 0, stream>>>(
        Xn, Wqm, nullptr, nullptr, nullptr, Qb, DD, DD, SCALE_Q);   // Q (scale folded)
    gemm_wmma<false, false, false, false, true><<<gN1, bW, 0, stream>>>(
        Xn, Wkm, nullptr, nullptr, nullptr, Kbf, DD, DD, 1.0f);     // K
    gemm_wmma<false, false, false, false, true><<<gN1, bW, 0, stream>>>(
        Xn, Wvm, nullptr, nullptr, nullptr, Vb, DD, DD, 1.0f);      // V
    transpose_v<<<dim3(TT / 32, BB * HH), bConv, 0, stream>>>(Vb, Vtb);
    attn_wmma<true><<<gAtt, bW, 0, stream>>>(Qb, Kbf, Vtb, Ob);
    gemm_wmma<true, false, true, true, true><<<gN1, bW, 0, stream>>>(
        Ob, Pm, pm_b, src, moutF, moutB, DD, DD, 1.0f);             // m_out = src + O@Pm + b

    // ---- final (cross) attention block ----
    rmsnorm_kernel<<<MTOT, 256, 0, stream>>>(att, nw, Xn);
    gemm_wmma<true, false, false, false, true><<<gN1, bW, 0, stream>>>(
        Xn, Wqf, bq_f, nullptr, nullptr, Qb, DD, DD, SCALE_Q);      // Qf = (An@W+b)*scale
    gemm_wmma<true, false, false, false, true><<<gN1, bW, 0, stream>>>(
        Xn, Wkf, bk_f, nullptr, nullptr, Kbf, DD, DD, 1.0f);        // Kf
    gemm_wmma<true, false, false, false, true><<<gN1, bW, 0, stream>>>(
        moutB, Wvf, bv_f, nullptr, nullptr, Vb, DD, DD, 1.0f);      // Vf from m_out
    transpose_v<<<dim3(TT / 32, BB * HH), bConv, 0, stream>>>(Vb, Vtb);
    attn_wmma<false><<<gAtt, bW, 0, stream>>>(Qb, Kbf, Vtb, Ob);
    gemm_wmma<true, false, true, true, false><<<gN1, bW, 0, stream>>>(
        Ob, Pf, pf_b, moutF, foutF, nullptr, DD, DD, 1.0f);         // f_out

    // ---- FFN block ----
    rmsnorm_kernel<<<MTOT, 256, 0, stream>>>(foutF, nw, Xn);
    gemm_wmma<true, true, false, false, true><<<gN4, bW, 0, stream>>>(
        Xn, W1t, b1, nullptr, nullptr, H1, 4 * DD, DD, 1.0f);       // gelu(h@w1+b1)
    gemm_wmma<true, false, true, true, false><<<gN1, bW, 0, stream>>>(
        H1, W2t, b2, foutF, out, nullptr, DD, 4 * DD, 1.0f);        // out = f_out + .@w2+b2
}
